// FeatureSimilarity_18416819765837
// MI455X (gfx1250) — matrix operand order repaired and checked by reading, not compiled, not run on hardware
//
#include <hip/hip_runtime.h>
#include <math.h>

typedef __attribute__((ext_vector_type(16))) __bf16 v16bf;
typedef __attribute__((ext_vector_type(8)))  __bf16 v8bf;
typedef __attribute__((ext_vector_type(8)))  float  v8f;
typedef __attribute__((ext_vector_type(4)))  float  f32x4;

// ---------------------------------------------------------------------------
// Kernel 1: one-time fp32 -> bf16 hi/lo split + exact fp32 row norms.
// ---------------------------------------------------------------------------
__global__ __launch_bounds__(256)
void split_rows(const float* __restrict__ F,
                __bf16* __restrict__ Fhi,
                __bf16* __restrict__ Flo,
                float* __restrict__ sq,
                int D) {
  __shared__ float red[256];
  const int row = blockIdx.x;
  const int t   = threadIdx.x;
  const float* src = F  + (size_t)row * D;
  __bf16* hi       = Fhi + (size_t)row * D;
  __bf16* lo       = Flo + (size_t)row * D;

  float acc = 0.0f;
  for (int k = t; k < D; k += 256) {
    float x = src[k];
    __bf16 h = (__bf16)x;
    hi[k] = h;
    lo[k] = (__bf16)(x - (float)h);
    acc += x * x;
  }
  red[t] = acc;
  __syncthreads();
  for (int s = 128; s > 0; s >>= 1) {
    if (t < s) red[t] += red[t + s];
    __syncthreads();
  }
  if (t == 0) sq[row] = red[0];
}

// ---------------------------------------------------------------------------
// Kernel 2 (fast path): upper-triangle 128x128 tiles, mirrored on write.
// 8 waves/block, each a 32x64 sub-tile = 2x4 WMMA 16x16x32 bf16 tiles.
// bf16x3 (hh + hl + lh) accumulated in fp32.
// ---------------------------------------------------------------------------
__global__ __launch_bounds__(256)
void l2_gemm_bf16x3_sym(const __bf16* __restrict__ Fhi,
                        const __bf16* __restrict__ Flo,
                        const float* __restrict__ sq,
                        float* __restrict__ out,
                        int N, int D, int T) {
  // Decode linear block index -> upper-triangle (bi, bj), bi <= bj.
  int rem = blockIdx.x;
  int bi = 0;
  int rowlen = T;
  while (rem >= rowlen) { rem -= rowlen; --rowlen; ++bi; }
  const int bj = bi + rem;

  const int I   = bi * 128;
  const int J   = bj * 128;
  const int tid = threadIdx.x;

  __shared__ float sqI[128];
  __shared__ float sqJ[128];
  if (tid < 128) sqI[tid] = sq[I + tid];
  else           sqJ[tid - 128] = sq[J + tid - 128];
  __syncthreads();

  const int wave  = tid >> 5;
  const int lane  = tid & 31;
  const int half  = lane >> 4;   // 0: lanes 0-15, 1: lanes 16-31
  const int l16   = lane & 15;
  const int m_off = (wave & 3) * 32;   // 4 waves along i -> 128 rows
  const int n_off = (wave >> 2) * 64;  // 2 waves along j -> 128 cols

  v8f acc[2][4];
#pragma unroll
  for (int mt = 0; mt < 2; ++mt)
#pragma unroll
    for (int nt = 0; nt < 4; ++nt) {
      v8f z = {};
      acc[mt][nt] = z;
    }

  const __bf16* aHi[2]; const __bf16* aLo[2];
  const __bf16* bHi[4]; const __bf16* bLo[4];
#pragma unroll
  for (int mt = 0; mt < 2; ++mt) {
    const size_t r = (size_t)(I + m_off + mt * 16 + l16) * D;
    aHi[mt] = Fhi + r;
    aLo[mt] = Flo + r;
  }
#pragma unroll
  for (int nt = 0; nt < 4; ++nt) {
    const size_t r = (size_t)(J + n_off + nt * 16 + l16) * D;
    bHi[nt] = Fhi + r;
    bLo[nt] = Flo + r;
  }

  for (int k0 = 0; k0 < D; k0 += 32) {
    // A fragment (16x32): half 0 -> K {k0+0..15},
    //                     half 1 -> K {k0+16..31}
    v16bf ahi[2], alo[2];
#pragma unroll
    for (int mt = 0; mt < 2; ++mt) {
      const v8bf h0 = *(const v8bf*)(aHi[mt] + k0 + half * 16);
      const v8bf h1 = *(const v8bf*)(aHi[mt] + k0 + 8 + half * 16);
      const v8bf l0 = *(const v8bf*)(aLo[mt] + k0 + half * 16);
      const v8bf l1 = *(const v8bf*)(aLo[mt] + k0 + 8 + half * 16);
      ahi[mt] = __builtin_shufflevector(h0, h1, 0,1,2,3,4,5,6,7,8,9,10,11,12,13,14,15);
      alo[mt] = __builtin_shufflevector(l0, l1, 0,1,2,3,4,5,6,7,8,9,10,11,12,13,14,15);
    }

    // B fragment (32x16): lane = column, half selects K {k0..15}/{k0+16..31}.
    v16bf bhi[4], blo[4];
#pragma unroll
    for (int nt = 0; nt < 4; ++nt) {
      bhi[nt] = *(const v16bf*)(bHi[nt] + k0 + half * 16);
      blo[nt] = *(const v16bf*)(bLo[nt] + k0 + half * 16);
    }

#pragma unroll
    for (int mt = 0; mt < 2; ++mt)
#pragma unroll
      for (int nt = 0; nt < 4; ++nt) {
        acc[mt][nt] = __builtin_amdgcn_wmma_f32_16x16x32_bf16(
            false, ahi[mt], false, bhi[nt], (short)0, acc[mt][nt], false, false);
        acc[mt][nt] = __builtin_amdgcn_wmma_f32_16x16x32_bf16(
            false, ahi[mt], false, blo[nt], (short)0, acc[mt][nt], false, false);
        acc[mt][nt] = __builtin_amdgcn_wmma_f32_16x16x32_bf16(
            false, alo[mt], false, bhi[nt], (short)0, acc[mt][nt], false, false);
      }
  }

  // Epilogue. C/D layout: VGPR r -> row M = r + 8*half, col N = lane&15.
  const bool mirror = (bi != bj);
#pragma unroll
  for (int mt = 0; mt < 2; ++mt) {
#pragma unroll
    for (int nt = 0; nt < 4; ++nt) {
      const int colL = n_off + nt * 16 + l16;
      const int gCol = J + colL;
      const float sj = sqJ[colL];
      float v[8];
#pragma unroll
      for (int r = 0; r < 8; ++r) {
        const int rowL = m_off + mt * 16 + r + half * 8;
        const int gRow = I + rowL;
        const float si = sqI[rowL];
        const float g = acc[mt][nt][r];
        float d2 = si + sj - 2.0f * g;
        d2 = d2 > 0.0f ? d2 : 0.0f;
        v[r] = (gRow == gCol) ? 0.0f : -sqrtf(d2);
        out[(size_t)gRow * N + gCol] = v[r];
      }
      if (mirror) {
        // Transposed tile: this lane owns output row gCol, 8 consecutive
        // columns starting at I + m_off + mt*16 + 8*half -> two b128 stores.
        const int c0 = I + m_off + mt * 16 + half * 8;
        float* p = out + (size_t)gCol * N + c0;
        f32x4 a = { v[0], v[1], v[2], v[3] };
        f32x4 b = { v[4], v[5], v[6], v[7] };
        *(f32x4*)(p)     = a;
        *(f32x4*)(p + 4) = b;
      }
    }
  }
}

// ---------------------------------------------------------------------------
// Fallback fused kernel (no workspace): in-register bf16 split per tile.
// ---------------------------------------------------------------------------
__device__ __forceinline__ void split4(f32x4 v, v16bf& hi, v16bf& lo, int base) {
#pragma unroll
  for (int e = 0; e < 4; ++e) {
    float x = v[e];
    __bf16 h = (__bf16)x;
    hi[base + e] = h;
    lo[base + e] = (__bf16)(x - (float)h);
  }
}

__global__ __launch_bounds__(256)
void l2_similarity_fused(const float* __restrict__ F,
                         float* __restrict__ out,
                         int N, int D) {
  __shared__ float sqI[128];
  __shared__ float sqJ[64];

  const int I   = blockIdx.x * 128;
  const int J   = blockIdx.y * 64;
  const int tid = threadIdx.x;

  if (tid < 192) {
    const int row = (tid < 128) ? (I + tid) : (J + tid - 128);
    const f32x4* p = (const f32x4*)(F + (size_t)row * D);
    float acc = 0.0f;
    const int kq = D >> 2;
    for (int k = 0; k < kq; ++k) {
      f32x4 v = p[k];
      acc += v[0] * v[0] + v[1] * v[1] + v[2] * v[2] + v[3] * v[3];
    }
    if (tid < 128) sqI[tid] = acc;
    else           sqJ[tid - 128] = acc;
  }
  __syncthreads();

  const int wave  = tid >> 5;
  const int lane  = tid & 31;
  const int half  = lane >> 4;
  const int l16   = lane & 15;
  const int m_off = (wave & 3) * 32;
  const int n_off = (wave >> 2) * 32;

  v8f acc[2][2];
#pragma unroll
  for (int mt = 0; mt < 2; ++mt)
#pragma unroll
    for (int nt = 0; nt < 2; ++nt) {
      v8f z = {};
      acc[mt][nt] = z;
    }

  const float* aRow[2];
  const float* bRow[2];
  aRow[0] = F + (size_t)(I + m_off + l16) * D;
  aRow[1] = aRow[0] + (size_t)16 * D;
  bRow[0] = F + (size_t)(J + n_off + l16) * D;
  bRow[1] = bRow[0] + (size_t)16 * D;

  for (int k0 = 0; k0 < D; k0 += 32) {
    v16bf ahi[2], alo[2], bhi[2], blo[2];
#pragma unroll
    for (int mt = 0; mt < 2; ++mt) {
      const float* r = aRow[mt];
      const f32x4* p0 = (const f32x4*)(r + k0 + half * 16);
      const f32x4* p1 = (const f32x4*)(r + k0 + 8 + half * 16);
      f32x4 v0 = p0[0], v1 = p0[1], v2 = p1[0], v3 = p1[1];
      split4(v0, ahi[mt], alo[mt], 0);
      split4(v1, ahi[mt], alo[mt], 4);
      split4(v2, ahi[mt], alo[mt], 8);
      split4(v3, ahi[mt], alo[mt], 12);
    }
#pragma unroll
    for (int nt = 0; nt < 2; ++nt) {
      const float* r = bRow[nt];
      const f32x4* p = (const f32x4*)(r + k0 + half * 16);
      f32x4 v0 = p[0], v1 = p[1], v2 = p[2], v3 = p[3];
      split4(v0, bhi[nt], blo[nt], 0);
      split4(v1, bhi[nt], blo[nt], 4);
      split4(v2, bhi[nt], blo[nt], 8);
      split4(v3, bhi[nt], blo[nt], 12);
    }
#pragma unroll
    for (int mt = 0; mt < 2; ++mt)
#pragma unroll
      for (int nt = 0; nt < 2; ++nt) {
        acc[mt][nt] = __builtin_amdgcn_wmma_f32_16x16x32_bf16(
            false, ahi[mt], false, bhi[nt], (short)0, acc[mt][nt], false, false);
        acc[mt][nt] = __builtin_amdgcn_wmma_f32_16x16x32_bf16(
            false, ahi[mt], false, blo[nt], (short)0, acc[mt][nt], false, false);
        acc[mt][nt] = __builtin_amdgcn_wmma_f32_16x16x32_bf16(
            false, alo[mt], false, bhi[nt], (short)0, acc[mt][nt], false, false);
      }
  }

#pragma unroll
  for (int mt = 0; mt < 2; ++mt) {
#pragma unroll
    for (int nt = 0; nt < 2; ++nt) {
      const int colL = n_off + nt * 16 + l16;
      const int gCol = J + colL;
      const float sj = sqJ[colL];
#pragma unroll
      for (int r = 0; r < 8; ++r) {
        const int rowL = m_off + mt * 16 + r + half * 8;
        const int gRow = I + rowL;
        const float si = sqI[rowL];
        const float g = acc[mt][nt][r];
        float d2 = si + sj - 2.0f * g;
        d2 = d2 > 0.0f ? d2 : 0.0f;
        const float val = (gRow == gCol) ? 0.0f : -sqrtf(d2);
        out[(size_t)gRow * N + gCol] = val;
      }
    }
  }
}

// ---------------------------------------------------------------------------
extern "C" void kernel_launch(void* const* d_in, const int* in_sizes, int n_in,
                              void* d_out, int out_size, void* d_ws, size_t ws_size,
                              hipStream_t stream) {
  (void)n_in;
  const float* F = (const float*)d_in[0];
  float* out = (float*)d_out;

  // out is N x N; features is N x D.
  int N = (int)(sqrt((double)out_size) + 0.5);
  while ((long long)N * N < (long long)out_size) ++N;
  while ((long long)N * N > (long long)out_size) --N;
  const int D = in_sizes[0] / N;

  // Workspace layout: Fhi[N*D] bf16 | Flo[N*D] bf16 | sq[N] fp32
  const size_t need = (size_t)N * D * 2 * 2 + (size_t)N * 4;

  if (d_ws != nullptr && ws_size >= need && (N % 128) == 0 && (D % 32) == 0) {
    __bf16* Fhi = (__bf16*)d_ws;
    __bf16* Flo = Fhi + (size_t)N * D;
    float*  sq  = (float*)(Flo + (size_t)N * D);

    split_rows<<<dim3(N), dim3(256), 0, stream>>>(F, Fhi, Flo, sq, D);

    const int T = N / 128;
    const int nblocks = T * (T + 1) / 2;  // upper-triangle tiles only
    l2_gemm_bf16x3_sym<<<dim3(nblocks), dim3(256), 0, stream>>>(
        Fhi, Flo, sq, out, N, D, T);
  } else {
    l2_similarity_fused<<<dim3(N / 128, N / 64), dim3(256), 0, stream>>>(
        F, out, N, D);
  }
}
